// SpectralSynthesis_31215822307555
// MI455X (gfx1250) — compile-verified, matrix operand order checked by reading, not run on hardware
//
#include <hip/hip_runtime.h>
#include <hip/hip_bf16.h>

typedef __attribute__((ext_vector_type(16))) _Float16 v16h;
typedef __attribute__((ext_vector_type(8)))  _Float16 v8h;
typedef __attribute__((ext_vector_type(8)))  float    v8f;

#define HDIM 64
#define HID  256
#define MROWS 2048
#define VSZ  32000

// ---------------------------------------------------------------------------
// Fragment loader for v_wmma_f32_16x16x32_f16 (works for global or LDS src).
// A 16x32 (ISA 7.12.2): lanes 0-15 row=lane, halves0-7=K0-7, halves8-15=K16-23;
// lanes 16-31 row=lane-16, halves0-7=K8-15, halves8-15=K24-31.  B mirrors it.
// ---------------------------------------------------------------------------
__device__ __forceinline__ v16h frag_load(const _Float16* M, int stride,
                                          int row_base, int k_base, int lane) {
  int r  = lane & 15;
  int ks = (lane >> 4) << 3;  // 0 or 8
  const _Float16* p = M + (size_t)(row_base + r) * stride + k_base + ks;
  union { v16h v; v8h h[2]; } u;
  u.h[0] = *(const v8h*)(p);
  u.h[1] = *(const v8h*)(p + 16);
  return u.v;
}

// 16-byte global -> LDS async copy (ASYNCcnt-tracked, CDNA5).
__device__ __forceinline__ void async_ld16(void* lds, const void* gl) {
  asm volatile("global_load_async_to_lds_b128 %0, %1, off"
               :: "v"((unsigned)(size_t)lds), "v"(gl)
               : "memory");
}
__device__ __forceinline__ void wait_async_le4() {
#if __has_builtin(__builtin_amdgcn_s_wait_asynccnt)
  __builtin_amdgcn_s_wait_asynccnt(4);
#else
  asm volatile("s_wait_asynccnt 0x4" ::: "memory");
#endif
}
__device__ __forceinline__ void wait_async_le0() {
#if __has_builtin(__builtin_amdgcn_s_wait_asynccnt)
  __builtin_amdgcn_s_wait_asynccnt(0);
#else
  asm volatile("s_wait_asynccnt 0x0" ::: "memory");
#endif
}

__device__ __forceinline__ float gelu_exact(float x) {
  return 0.5f * x * (1.0f + erff(x * 0.70710678118654752f));
}

// ---------------------------------------------------------------------------
// Kernel 1: per-row prep.  8 rows per 256-thread block.
// ---------------------------------------------------------------------------
__global__ void __launch_bounds__(256) prep_rows(
    const float* __restrict__ wrep,
    const float* __restrict__ W0, const float* __restrict__ b0,
    const float* __restrict__ W1, const float* __restrict__ b1,
    const float* __restrict__ W2, const float* __restrict__ b2,
    const float* __restrict__ rw,
    _Float16* __restrict__ xA, _Float16* __restrict__ phA,
    _Float16* __restrict__ fA, float* __restrict__ f2, float* __restrict__ la) {
  __shared__ float wr[8][192];
  __shared__ float xs[8][256];
  __shared__ float xs2[8][256];
  int t  = threadIdx.x;
  int m0 = blockIdx.x * 8;

  for (int idx = t; idx < 8 * 192; idx += 256) {
    int r = idx / 192, k = idx % 192;
    wr[r][k] = wrep[(size_t)(m0 + r) * 192 + k];
  }
  __syncthreads();

  {  // stats: one wave per row
    int r = t >> 5, l = t & 31;
    float a0 = wr[r][l], a1 = wr[r][l + 32];
    float s2 = a0 * a0 + a1 * a1;
    float sa = wr[r][64 + l] + wr[r][96 + l];
    for (int off = 16; off > 0; off >>= 1) {
      s2 += __shfl_down(s2, off);
      sa += __shfl_down(sa, off);
    }
    if (l == 0) { f2[m0 + r] = s2; la[m0 + r] = log1pf(sa); }
  }

  for (int idx = t; idx < 8 * 64; idx += 256) {
    int r = idx >> 6, h = idx & 63;
    float ph = wr[r][128 + h];
    phA[(size_t)(m0 + r) * 128 + h]      = (_Float16)cosf(ph);
    phA[(size_t)(m0 + r) * 128 + 64 + h] = (_Float16)sinf(ph);
    fA[(size_t)(m0 + r) * 64 + h]        = (_Float16)wr[r][h];
  }

  float acc[8];
  float bb = b0[t];
  for (int r = 0; r < 8; ++r) acc[r] = bb;
  for (int k = 0; k < 192; ++k) {
    float w = W0[k * 256 + t];
#pragma unroll
    for (int r = 0; r < 8; ++r) acc[r] = fmaf(wr[r][k], w, acc[r]);
  }
  for (int r = 0; r < 8; ++r) xs[r][t] = gelu_exact(acc[r]);
  __syncthreads();

  bb = b1[t];
  for (int r = 0; r < 8; ++r) acc[r] = bb;
  for (int k = 0; k < 256; ++k) {
    float w = W1[k * 256 + t];
#pragma unroll
    for (int r = 0; r < 8; ++r) acc[r] = fmaf(xs[r][k], w, acc[r]);
  }
  float rw1 = rw[1], rw2 = rw[2];
  for (int r = 0; r < 8; ++r) xs2[r][t] = gelu_exact(acc[r]) + rw1 * xs[r][t];
  __syncthreads();

  bb = b2[t];
  for (int r = 0; r < 8; ++r) acc[r] = bb;
  for (int k = 0; k < 256; ++k) {
    float w = W2[k * 256 + t];
#pragma unroll
    for (int r = 0; r < 8; ++r) acc[r] = fmaf(xs2[r][k], w, acc[r]);
  }
  for (int r = 0; r < 8; ++r)
    xA[(size_t)(m0 + r) * 256 + t] = (_Float16)(gelu_exact(acc[r]) + rw2 * xs2[r][t]);
}

// ---------------------------------------------------------------------------
// Kernel 2a: vocab-side prep. 8 vocab rows per block (one wave each).
// ---------------------------------------------------------------------------
__global__ void __launch_bounds__(256) prep_vocab(
    const float* __restrict__ vf, const float* __restrict__ vp,
    _Float16* __restrict__ vfB, _Float16* __restrict__ phB,
    float* __restrict__ v2) {
  int t = threadIdx.x;
  int r = t >> 5, l = t & 31;
  int v = blockIdx.x * 8 + r;
  float a0 = vf[(size_t)v * 64 + l], a1 = vf[(size_t)v * 64 + 32 + l];
  vfB[(size_t)v * 64 + l]      = (_Float16)a0;
  vfB[(size_t)v * 64 + 32 + l] = (_Float16)a1;
  float p0 = vp[(size_t)v * 64 + l], p1 = vp[(size_t)v * 64 + 32 + l];
  phB[(size_t)v * 128 + l]      = (_Float16)cosf(p0);
  phB[(size_t)v * 128 + 32 + l] = (_Float16)cosf(p1);
  phB[(size_t)v * 128 + 64 + l] = (_Float16)sinf(p0);
  phB[(size_t)v * 128 + 96 + l] = (_Float16)sinf(p1);
  float s = a0 * a0 + a1 * a1;
  for (int off = 16; off > 0; off >>= 1) s += __shfl_down(s, off);
  if (l == 0) v2[v] = s;
}

// ---------------------------------------------------------------------------
// Kernel 2b: Wv (256 x V, k-major) -> wvT (V x 256, f16) via LDS tile.
// ---------------------------------------------------------------------------
__global__ void __launch_bounds__(256) prep_wvT(const float* __restrict__ Wv,
                                                _Float16* __restrict__ wvT) {
  __shared__ float tile[64][65];
  int t  = threadIdx.x;
  int v0 = blockIdx.x * 64;
  int k0 = blockIdx.y * 64;
#pragma unroll
  for (int i = 0; i < 16; ++i) {
    int idx = i * 256 + t;
    int kl = idx >> 6, vl = idx & 63;
    tile[kl][vl] = Wv[(size_t)(k0 + kl) * VSZ + v0 + vl];
  }
  __syncthreads();
  int vl = t >> 2;
  int kc = (t & 3) << 4;
  union { v8h h; _Float16 e[8]; } u0, u1;
#pragma unroll
  for (int j = 0; j < 8; ++j) {
    u0.e[j] = (_Float16)tile[kc + j][vl];
    u1.e[j] = (_Float16)tile[kc + 8 + j][vl];
  }
  _Float16* dst = wvT + (size_t)(v0 + vl) * 256 + k0 + kc;
  *(v8h*)dst       = u0.h;
  *(v8h*)(dst + 8) = u1.h;
}

// ---------------------------------------------------------------------------
// Kernel 3: fused spectral GEMM with async double-buffered LDS B-panels.
// Block = 8 waves x 16 rows = 128 rows; sweeps 25 16-col tiles.
// Per tile: 4 async b128 copies/thread (14KB panel, one copy per BLOCK),
// s_wait_asynccnt + barrier, then 14 v_wmma_f32_16x16x32_f16 fed from LDS.
// ---------------------------------------------------------------------------
__global__ void __launch_bounds__(256) spectral_main(
    const _Float16* __restrict__ fA, const _Float16* __restrict__ phA,
    const _Float16* __restrict__ xA, const _Float16* __restrict__ vfB,
    const _Float16* __restrict__ phB, const _Float16* __restrict__ wvT,
    const float* __restrict__ f2, const float* __restrict__ la,
    const float* __restrict__ v2, const float* __restrict__ bv,
    float* __restrict__ out) {
  __shared__ __align__(16) _Float16 lvf[2][16 * 64];
  __shared__ __align__(16) _Float16 lph[2][16 * 128];
  __shared__ __align__(16) _Float16 lwv[2][16 * 256];

  int tid   = threadIdx.x;
  int lane  = tid & 31;
  int wave  = tid >> 5;
  int m_base = blockIdx.y * 128 + wave * 16;
  int tile0  = blockIdx.x * 25;

  // A-side fragments, register-resident across the N sweep.
  v16h aF[2], aP[4], aX[8];
#pragma unroll
  for (int j = 0; j < 2; ++j) aF[j] = frag_load(fA, 64, m_base, 32 * j, lane);
#pragma unroll
  for (int j = 0; j < 4; ++j) aP[j] = frag_load(phA, 128, m_base, 32 * j, lane);
#pragma unroll
  for (int j = 0; j < 8; ++j) aX[j] = frag_load(xA, 256, m_base, 32 * j, lane);

  int m_off = (lane >> 4) << 3;
  float f2r[8], lar[8];
#pragma unroll
  for (int i = 0; i < 8; ++i) {
    f2r[i] = f2[m_base + m_off + i];
    lar[i] = la[m_base + m_off + i];
  }

  // Issue one shared async copy of the 14KB B-panel (4 ops per thread, so
  // every wave's ASYNCcnt advances uniformly; the 2KB vf slab is issued
  // twice -- identical data to identical LDS bytes -- to keep it uniform).
  auto issue_copy = [&](int tile, int buf) {
    int nb = tile * 16;
    const _Float16* gvf = vfB + (size_t)nb * 64;
    const _Float16* gph = phB + (size_t)nb * 128;
    const _Float16* gwv = wvT + (size_t)nb * 256;
    int c0 = (tid & 127) * 8;
    async_ld16(&lvf[buf][c0],             gvf + c0);
    async_ld16(&lph[buf][tid * 8],        gph + tid * 8);
    async_ld16(&lwv[buf][tid * 8],        gwv + tid * 8);
    async_ld16(&lwv[buf][(tid + 256) * 8], gwv + (tid + 256) * 8);
  };

  issue_copy(tile0, 0);

  for (int t = 0; t < 25; ++t) {
    int buf = t & 1;
    if (t + 1 < 25) {
      issue_copy(tile0 + t + 1, buf ^ 1);
      wait_async_le4();   // copy(t) complete; copy(t+1)'s 4 ops still in flight
    } else {
      wait_async_le0();
    }
    __syncthreads();      // all waves' panel writes visible

    if (wave == 0 && t + 2 < 25) {  // warm L2 for the tile after next
      const char* pf = (const char*)(wvT + (size_t)(tile0 + t + 2) * 16 * 256);
      __builtin_prefetch(pf + lane * 256, 0, 1);
    }

    v8f cC = {}, cP = {}, cS = {};
#pragma unroll
    for (int j = 0; j < 2; ++j) {
      v16h b = frag_load(&lvf[buf][0], 64, 0, 32 * j, lane);
      cC = __builtin_amdgcn_wmma_f32_16x16x32_f16(false, aF[j], false, b,
                                                  (short)0, cC, false, false);
    }
#pragma unroll
    for (int j = 0; j < 4; ++j) {
      v16h b = frag_load(&lph[buf][0], 128, 0, 32 * j, lane);
      cP = __builtin_amdgcn_wmma_f32_16x16x32_f16(false, aP[j], false, b,
                                                  (short)0, cP, false, false);
    }
#pragma unroll
    for (int j = 0; j < 8; ++j) {
      v16h b = frag_load(&lwv[buf][0], 256, 0, 32 * j, lane);
      cS = __builtin_amdgcn_wmma_f32_16x16x32_f16(false, aX[j], false, b,
                                                  (short)0, cS, false, false);
    }

    int n_base = (tile0 + t) * 16;
    int n = n_base + (lane & 15);
    float v2n = v2[n], bvn = bv[n];
    size_t rowbase = (size_t)(m_base + m_off) * VSZ + n;
#pragma unroll
    for (int i = 0; i < 8; ++i) {
      float d  = fmaxf(f2r[i] + v2n - 2.0f * cC[i], 0.0f);
      float pm = (-sqrtf(d) + 0.5f * (cP[i] * (1.0f / 64.0f))) * lar[i];
      out[rowbase + (size_t)i * VSZ] = pm + 0.1f * (cS[i] + bvn);
    }
    __syncthreads();      // reads of buf done before it is overwritten
  }
}

// ---------------------------------------------------------------------------
extern "C" void kernel_launch(void* const* d_in, const int* in_sizes, int n_in,
                              void* d_out, int out_size, void* d_ws,
                              size_t ws_size, hipStream_t stream) {
  const float* wrep = (const float*)d_in[0];
  const float* vf   = (const float*)d_in[1];
  const float* vp   = (const float*)d_in[2];
  const float* W0   = (const float*)d_in[3];
  const float* b0   = (const float*)d_in[4];
  const float* W1   = (const float*)d_in[5];
  const float* b1   = (const float*)d_in[6];
  const float* W2   = (const float*)d_in[7];
  const float* b2   = (const float*)d_in[8];
  const float* Wv   = (const float*)d_in[9];
  const float* bv   = (const float*)d_in[10];
  const float* rw   = (const float*)d_in[11];
  float* out = (float*)d_out;

  char* ws = (char*)d_ws;
  size_t off = 0;
  auto alloc = [&](size_t bytes) -> char* {
    char* p = ws + off;
    off = (off + bytes + 255) & ~(size_t)255;
    return p;
  };
  _Float16* xA  = (_Float16*)alloc((size_t)MROWS * HID * 2);
  _Float16* phA = (_Float16*)alloc((size_t)MROWS * 128 * 2);
  _Float16* fA  = (_Float16*)alloc((size_t)MROWS * HDIM * 2);
  float*    f2  = (float*)alloc((size_t)MROWS * 4);
  float*    la  = (float*)alloc((size_t)MROWS * 4);
  _Float16* vfB = (_Float16*)alloc((size_t)VSZ * HDIM * 2);
  _Float16* phB = (_Float16*)alloc((size_t)VSZ * 128 * 2);
  _Float16* wvT = (_Float16*)alloc((size_t)VSZ * HID * 2);
  float*    v2  = (float*)alloc((size_t)VSZ * 4);

  prep_rows<<<MROWS / 8, 256, 0, stream>>>(wrep, W0, b0, W1, b1, W2, b2, rw,
                                           xA, phA, fA, f2, la);
  prep_vocab<<<VSZ / 8, 256, 0, stream>>>(vf, vp, vfB, phB, v2);
  prep_wvT<<<dim3(VSZ / 64, HID / 64), 256, 0, stream>>>(Wv, wvT);
  spectral_main<<<dim3(80, 16), 256, 0, stream>>>(fA, phA, xA, vfB, phB, wvT,
                                                  f2, la, v2, bv, out);
}
